// ConvBlockTemp_27015344292177
// MI455X (gfx1250) — matrix operand order repaired and checked by reading, not compiled, not run on hardware
//
#include <hip/hip_runtime.h>
#include <hip/hip_bf16.h>

// ---------------------------------------------------------------------------
// ConvBlockTemp on MI455X (gfx1250): Chebyshev SpMM (f32 atomics) +
// temporal/channel contraction as bf16 WMMA GEMM (v_wmma_f32_16x16x32_bf16) +
// fused bias/BN-stats epilogue + in-place BN+ReLU.
// ---------------------------------------------------------------------------

typedef __attribute__((ext_vector_type(16))) __bf16 v16bf;
typedef __attribute__((ext_vector_type(8)))  __bf16 v8bf;
typedef __attribute__((ext_vector_type(4)))  __bf16 v4bf;
typedef __attribute__((ext_vector_type(8)))  float  v8f;

#define V_N    49152
#define B_N    4
#define T_N    8
#define FIN_N  32
#define FOUT_N 64
#define K_N    3
#define TW_N   3
#define NNZ_N  (9 * V_N)                 // 442368
#define COLS   (B_N * T_N * FIN_N)       // 1024
#define M_TOT  ((size_t)V_N * B_N * T_N) // 1572864
#define RED_N  (K_N * TW_N * FIN_N)      // 288
#define XK_ELEMS ((size_t)V_N * COLS)    // 50331648

// ---- workspace layout -----------------------------------------------------
// f32 region (float offsets from ws base):
//   x1    : [0, XK_ELEMS)
//   x2acc : [XK_ELEMS, 2*XK_ELEMS)
//   stats : [2*XK_ELEMS, 2*XK_ELEMS + 128)   (sum[64], sumsq[64])
// bf16 region (byte offset 402653696, 32B aligned):
//   xk0, xk1, xk2 : XK_ELEMS bf16 each
//   wtT[64][288]  : transposed bf16 weight (36864 B)
#define WS_BF16_BYTE_OFF 402653696ull

// ---------------------------------------------------------------------------
__global__ void k_zero(float* __restrict__ p, long long n4) {
  long long i = (long long)blockIdx.x * blockDim.x + threadIdx.x;
  if (i < n4) ((float4*)p)[i] = make_float4(0.f, 0.f, 0.f, 0.f);
}

// weight [K*TW*FIN=288][FOUT=64] f32  ->  wt bf16 transposed [64][288]
__global__ void k_wt(const float* __restrict__ w, __bf16* __restrict__ wt) {
  int i = blockIdx.x * blockDim.x + threadIdx.x;
  if (i < RED_N * FOUT_N) {
    int n = i / RED_N, r = i % RED_N;
    wt[i] = (__bf16)w[r * FOUT_N + n];
  }
}

// x1[row, c] += val * x0[col, c]   (x0 gathered straight from input layout)
__global__ void k_spmm1(const float* __restrict__ x, const float* __restrict__ vals,
                        const int* __restrict__ rows, const int* __restrict__ cols,
                        float* __restrict__ x1) {
  const int e = blockIdx.x;
  const float val = vals[e];
  const int row = rows[e], col = cols[e];
  const int c0 = threadIdx.x * 4;
  const int b = c0 >> 8, t = (c0 >> 5) & 7, f = c0 & 31;
  const float4 s = *(const float4*)&x[(((size_t)b * V_N + col) * T_N + t) * FIN_N + f];
  float* dst = x1 + (size_t)row * COLS + c0;
  atomicAdd(dst + 0, val * s.x);
  atomicAdd(dst + 1, val * s.y);
  atomicAdd(dst + 2, val * s.z);
  atomicAdd(dst + 3, val * s.w);
}

// x2acc[row, c] += 2*val * x1[col, c]
__global__ void k_spmm2(const float* __restrict__ x1, const float* __restrict__ vals,
                        const int* __restrict__ rows, const int* __restrict__ cols,
                        float* __restrict__ x2acc) {
  const int e = blockIdx.x;
  const float val = 2.0f * vals[e];
  const int row = rows[e], col = cols[e];
  const int c0 = threadIdx.x * 4;
  const float4 s = *(const float4*)&x1[(size_t)col * COLS + c0];
  float* dst = x2acc + (size_t)row * COLS + c0;
  atomicAdd(dst + 0, val * s.x);
  atomicAdd(dst + 1, val * s.y);
  atomicAdd(dst + 2, val * s.z);
  atomicAdd(dst + 3, val * s.w);
}

// xk0 = bf16(x0), xk1 = bf16(x1), xk2 = bf16(x2acc - x0)
__global__ void k_finalize(const float* __restrict__ x, const float* __restrict__ x1,
                           const float* __restrict__ x2acc,
                           __bf16* __restrict__ xk0, __bf16* __restrict__ xk1,
                           __bf16* __restrict__ xk2) {
  const size_t i = ((size_t)blockIdx.x * blockDim.x + threadIdx.x) * 4;
  const size_t v = i >> 10;
  const int c = (int)(i & 1023);
  const int b = c >> 8, t = (c >> 5) & 7, f = c & 31;
  const float4 x0v = *(const float4*)&x[(((size_t)b * V_N + v) * T_N + t) * FIN_N + f];
  const float4 a1  = *(const float4*)&x1[i];
  const float4 a2  = *(const float4*)&x2acc[i];
  v4bf o0 = {(__bf16)x0v.x, (__bf16)x0v.y, (__bf16)x0v.z, (__bf16)x0v.w};
  v4bf o1 = {(__bf16)a1.x, (__bf16)a1.y, (__bf16)a1.z, (__bf16)a1.w};
  v4bf o2 = {(__bf16)(a2.x - x0v.x), (__bf16)(a2.y - x0v.y),
             (__bf16)(a2.z - x0v.z), (__bf16)(a2.w - x0v.w)};
  *(v4bf*)&xk0[i] = o0;
  *(v4bf*)&xk1[i] = o1;
  *(v4bf*)&xk2[i] = o2;
}

// ---------------------------------------------------------------------------
// WMMA GEMM: out[m, n] = sum_{s=0..8} A_s[m, 0..31] . W_s[0..31, n]
//   m = (v*B + b)*T + t ; s = k*3 + tw ; A_s row = xk[k][v][b][t+tw-1][fin]
// One wave = one 16-row M tile x full N=64 (4 accumulators).
// ---------------------------------------------------------------------------
__global__ __launch_bounds__(256)
void k_gemm(const __bf16* __restrict__ xk0, const __bf16* __restrict__ xk1,
            const __bf16* __restrict__ xk2, const __bf16* __restrict__ wt,
            const float* __restrict__ bias, float* __restrict__ out,
            float* __restrict__ stats) {
  const __bf16* xk[3] = {xk0, xk1, xk2};
  const int wave = threadIdx.x >> 5;
  const int lane = threadIdx.x & 31;
  const int mtile = blockIdx.x * 8 + wave;
  const int m0 = mtile << 4;
  const int ml = lane & 15;      // A row within tile / C column within n-tile
  const int hi = lane >> 4;      // K sub-block selector per ISA layout
  const int m = m0 + ml;
  const int v = m >> 5, b = (m >> 3) & 3, t = m & 7;

  v8f acc[4] = {};

#pragma unroll
  for (int s = 0; s < 9; ++s) {
    const int k = s / 3, tw = s % 3;
    const int tp = t + tw - 1;           // temporal window (zero-padded)
    v16bf a = {};
    if (tp >= 0 && tp < T_N) {
      // A operand, contiguous K order used here: element e holds K = hi*16 + e
      const __bf16* ar = xk[k] + (((size_t)v * B_N + b) * T_N + tp) * FIN_N + hi * 16;
      const v8bf lo = *(const v8bf*)ar;          // fin hi*16 .. hi*16+7
      const v8bf hv = *(const v8bf*)(ar + 8);    // fin hi*16+8 .. hi*16+15
#pragma unroll
      for (int i2 = 0; i2 < 8; ++i2) { a[i2] = lo[i2]; a[8 + i2] = hv[i2]; }
    }
#pragma unroll
    for (int nt = 0; nt < 4; ++nt) {
      // B operand, same contiguous K order as A: lane -> N, element e holds K = hi*16 + e
      // wt is [n][r] so 16 consecutive K are contiguous (one 32B load).
      const v16bf bfrag =
          *(const v16bf*)(wt + (size_t)(nt * 16 + ml) * RED_N + s * 32 + hi * 16);
      acc[nt] = __builtin_amdgcn_wmma_f32_16x16x32_bf16(
          false, a, false, bfrag, (short)0, acc[nt], false, false);
    }
  }

  // Epilogue: bias, store y to d_out ([B,V,T,FOUT]), accumulate BN stats.
  // C/D layout: VGPR j -> row m0 + j + hi*8, column = ml (within n-tile).
#pragma unroll
  for (int nt = 0; nt < 4; ++nt) {
    const int n = nt * 16 + ml;
    const float bv = bias[n];
    float s1 = 0.f, s2 = 0.f;
#pragma unroll
    for (int j = 0; j < 8; ++j) {
      const int row = m0 + j + hi * 8;
      const int rv = row >> 5, rb = (row >> 3) & 3, rt = row & 7;
      const float y = acc[nt][j] + bv;
      out[(((size_t)rb * V_N + rv) * T_N + rt) * FOUT_N + n] = y;
      s1 += y;
      s2 += y * y;
    }
    atomicAdd(&stats[n], s1);
    atomicAdd(&stats[FOUT_N + n], s2);
  }
}

// In-place BatchNorm (training-mode batch stats) + ReLU on d_out.
__global__ void k_bn(float* __restrict__ y, const float* __restrict__ stats,
                     const float* __restrict__ gamma, const float* __restrict__ beta) {
  const size_t i = ((size_t)blockIdx.x * blockDim.x + threadIdx.x) * 4;
  const int c = (int)(i & 63);
  float4 vy = *(float4*)&y[i];
  const float invM = 1.0f / (float)M_TOT;
  float in[4] = {vy.x, vy.y, vy.z, vy.w};
  float o[4];
#pragma unroll
  for (int j = 0; j < 4; ++j) {
    const int ch = c + j;
    const float mean = stats[ch] * invM;
    const float var = stats[FOUT_N + ch] * invM - mean * mean;
    const float r = rsqrtf(var + 1e-5f);
    const float t = (in[j] - mean) * r * gamma[ch] + beta[ch];
    o[j] = t > 0.f ? t : 0.f;
  }
  *(float4*)&y[i] = make_float4(o[0], o[1], o[2], o[3]);
}

// ---------------------------------------------------------------------------
extern "C" void kernel_launch(void* const* d_in, const int* in_sizes, int n_in,
                              void* d_out, int out_size, void* d_ws, size_t ws_size,
                              hipStream_t stream) {
  const float* x     = (const float*)d_in[0];
  const float* vals  = (const float*)d_in[1];
  const float* w     = (const float*)d_in[2];
  const float* bias  = (const float*)d_in[3];
  const float* gamma = (const float*)d_in[4];
  const float* beta  = (const float*)d_in[5];
  const int*   rows  = (const int*)d_in[6];
  const int*   cols  = (const int*)d_in[7];
  float* out = (float*)d_out;

  float* f32ws = (float*)d_ws;
  float* x1    = f32ws;
  float* x2acc = f32ws + XK_ELEMS;
  float* stats = f32ws + 2 * XK_ELEMS;
  __bf16* bfb = (__bf16*)((char*)d_ws + WS_BF16_BYTE_OFF);
  __bf16* xk0 = bfb;
  __bf16* xk1 = bfb + XK_ELEMS;
  __bf16* xk2 = bfb + 2 * XK_ELEMS;
  __bf16* wt  = bfb + 3 * XK_ELEMS;

  // zero x1 + x2acc + stats (2*XK_ELEMS + 128 floats = 25165856 float4)
  const long long n4 = (long long)(2 * XK_ELEMS + 128) / 4;
  k_zero<<<(unsigned)((n4 + 255) / 256), 256, 0, stream>>>(f32ws, n4);
  k_wt<<<(RED_N * FOUT_N + 255) / 256, 256, 0, stream>>>(w, wt);
  k_spmm1<<<NNZ_N, 256, 0, stream>>>(x, vals, rows, cols, x1);
  k_spmm2<<<NNZ_N, 256, 0, stream>>>(x1, vals, rows, cols, x2acc);
  k_finalize<<<(unsigned)(XK_ELEMS / (256 * 4)), 256, 0, stream>>>(x, x1, x2acc,
                                                                  xk0, xk1, xk2);
  k_gemm<<<(unsigned)(M_TOT / 16 / 8), 256, 0, stream>>>(xk0, xk1, xk2, wt, bias,
                                                         out, stats);
  k_bn<<<(unsigned)(M_TOT * FOUT_N / (256 * 4)), 256, 0, stream>>>(out, stats,
                                                                   gamma, beta);
}
